// CFormerAdapter_79602923864482
// MI455X (gfx1250) — matrix operand order repaired and checked by reading, not compiled, not run on hardware
//
#include <hip/hip_runtime.h>
#include <hip/hip_bf16.h>

// CFormer forward for MI455X (gfx1250, wave32).
//   k_alphas      : sigmoid(last channel)*mask, per-batch sums
//   k_cif_events  : sequential CIF recurrence, 1 lane per batch in ONE wave;
//                   cross-batch "any ready" coupling via ballot_w32; emits sparse
//                   (t, w) events grouped by monotone token index.
//   k_cif_gather  : H[b,n,:] = sum over events of w * hidden[b,t,:]  (no atomics)
//   k_cvt_w       : fp32 -> bf16 weight conversion (K padded to 1024)
//   k_gemm_bf16   : 32x64 tile per wave, 8x v_wmma_f32_16x16x32_bf16 per k-step
//                   (2 A frags x 4 B frags -> ~21 FLOP/B from cache vs 12.8 at 16x64)
//   k_rmsnorm     : fp32 RMS over D, * ln_w, -> bf16 for GEMM2
//   k_tail        : num_audio_tokens / num_pred_audio_tokens outputs
//
// Roofline: ~63 GFLOP of GEMM, text_proj traffic ~140MB (output-dominated).
// bf16 WMMA (K=32/instr) is the throughput path; f32 WMMA is K=4/instr.

typedef __attribute__((ext_vector_type(16))) __bf16 v16bf;
typedef __attribute__((ext_vector_type(8)))  __bf16 v8bf;
typedef __attribute__((ext_vector_type(8)))  float  v8f;

#define EV_CAP 12288   // >= T * (1 set + 3 extra sets + 1 add) = 7500 worst case

// ---------------------------------------------------------------- alphas + sums
__global__ void __launch_bounds__(256) k_alphas(
    const float* __restrict__ audio, const int* __restrict__ num_frames,
    float* __restrict__ alphas, float* __restrict__ sums, int T, int D) {
  int b = blockIdx.x;
  int nf = num_frames[b];
  __shared__ float red[256];
  float acc = 0.f;
  for (int t = threadIdx.x; t < T; t += 256) {
    float x = audio[((size_t)b * T + t) * D + (D - 1)];
    float a = 1.f / (1.f + expf(-x));
    a = (t < nf) ? a : 0.f;
    alphas[(size_t)b * T + t] = a;
    acc += a;
  }
  red[threadIdx.x] = acc;
  __syncthreads();
  for (int s = 128; s > 0; s >>= 1) {
    if (threadIdx.x < s) red[threadIdx.x] += red[threadIdx.x + s];
    __syncthreads();
  }
  if (threadIdx.x == 0) sums[b] = red[0];
}

// --------------------------------------------------- CIF recurrence -> sparse events
__global__ void __launch_bounds__(32) k_cif_events(
    const float* __restrict__ alphas, const float* __restrict__ sums,
    const int* __restrict__ num_text_tokens,
    float* __restrict__ ev_w, int* __restrict__ ev_t, int* __restrict__ tok_start,
    int B, int T, int N) {
  int b = threadIdx.x;
  bool active = (b < B);
  const float* al = alphas + (size_t)b * T;
  float* ew = ev_w + (size_t)b * EV_CAP;
  int*   et = ev_t + (size_t)b * EV_CAP;
  int*   ts = tok_start + (size_t)b * (N + 1);

  int nt = 1;
  float scale = 0.f;
  if (active) {
    nt = num_text_tokens[b]; if (nt < 1) nt = 1;
    float s = sums[b];
    scale = (s != 0.f) ? (float)nt / s : 0.f;
    ts[0] = 0;
  }

  float integrate = 0.f, remainds = 0.f;
  int ti = 0, ep = 0, cur = 0;
  int last_set_t = -1, last_set_idx = -1, last_set_pos = -1;

  for (int t = 0; t < T; ++t) {
    // finalize column t-1: add previous remainder at current token_index
    if (active && t > 0 && remainds != 0.f) {
      while (cur < ti) ts[++cur] = ep;
      et[ep] = t - 1; ew[ep] = remainds; ++ep;
    }
    float alpha = active ? al[t] * scale : 0.f;
    float alpha_needed = 1.f - integrate;
    integrate += alpha;

    // unconditional first pass of the do-while body
    {
      bool ready = (integrate >= 1.f);
      if (ready) integrate -= 1.f;
      float alpha_int = ready ? alpha_needed : alpha;
      if (active) {
        if (t == last_set_t && ti == last_set_idx) {
          ew[last_set_pos] = alpha_int;                 // scatter .set overwrite
        } else {
          while (cur < ti) ts[++cur] = ep;
          et[ep] = t; ew[ep] = alpha_int;
          last_set_t = t; last_set_idx = ti; last_set_pos = ep; ++ep;
        }
      }
      remainds = alpha - alpha_int;
      if (ready) { ti = ti + 1; if (ti > nt - 1) ti = nt - 1; }
    }

    // bounded extra fires; reference fires for ALL batches if ANY is ready
    for (int p = 0; p < 3; ++p) {
      bool any_ready = (__builtin_amdgcn_ballot_w32(active && (integrate >= 1.f)) != 0u);
      if (!any_ready) continue;
      bool ready = (integrate >= 1.f);
      if (ready) integrate -= 1.f;
      float a2 = remainds;
      float alpha_int = ready ? 1.f : a2;
      if (active) {
        if (t == last_set_t && ti == last_set_idx) {
          ew[last_set_pos] = alpha_int;
        } else {
          while (cur < ti) ts[++cur] = ep;
          et[ep] = t; ew[ep] = alpha_int;
          last_set_t = t; last_set_idx = ti; last_set_pos = ep; ++ep;
        }
      }
      remainds = a2 - alpha_int;
      if (ready) { ti = ti + 1; if (ti > nt - 1) ti = nt - 1; }
    }
    // last column (t==T-1) never gets the trailing remainder add (matches ref)
  }
  if (active) { while (cur < N) ts[++cur] = ep; }   // ts[N] = ep
}

// ------------------------------------------- H[b,n,:] = sum_events w * hidden[b,t,:]
__global__ void __launch_bounds__(256) k_cif_gather(
    const float* __restrict__ audio, const float* __restrict__ ev_w,
    const int* __restrict__ ev_t, const int* __restrict__ tok_start,
    __bf16* __restrict__ Hbf, int T, int D, int N, int Kp) {
  int n = blockIdx.x, b = blockIdx.y;
  int Dm1 = D - 1;
  int s = tok_start[(size_t)b * (N + 1) + n];
  int e = tok_start[(size_t)b * (N + 1) + n + 1];
  const float* ew = ev_w + (size_t)b * EV_CAP;
  const int*   et = ev_t + (size_t)b * EV_CAP;
  size_t row = ((size_t)b * N + n) * Kp;
  for (int d = threadIdx.x; d < Kp; d += 256) {
    float acc = 0.f;
    if (d < Dm1) {
      for (int i = s; i < e; ++i) {
        acc += ew[i] * audio[((size_t)b * T + et[i]) * D + d];
      }
    }
    Hbf[row + d] = (__bf16)acc;   // pad column (d==Dm1..Kp-1) written as 0
  }
}

// ----------------------------------------------------------- weight fp32 -> bf16 (pad K)
__global__ void __launch_bounds__(256) k_cvt_w(
    const float* __restrict__ W, __bf16* __restrict__ Wb, int rows, int K, int Kp) {
  size_t idx = (size_t)blockIdx.x * 256 + threadIdx.x;
  size_t total = (size_t)rows * Kp;
  if (idx >= total) return;
  int r = (int)(idx / Kp), k = (int)(idx % Kp);
  Wb[idx] = (k < K) ? (__bf16)W[(size_t)r * K + k] : (__bf16)0.f;
}

// ------------------------------------------------------------------ zero filler
__global__ void __launch_bounds__(256) k_zero_bf16(__bf16* __restrict__ p, size_t n) {
  size_t i = (size_t)blockIdx.x * 256 + threadIdx.x;
  if (i < n) p[i] = (__bf16)0.f;
}

// --------------------------------------------------------------- WMMA bf16 GEMM
// A fragment layout (16x32 bf16, wave32): lane m = lane&15, K-halves by lane>>4:
//   elems 0..7  -> K = 8*(lane>>4) + 0..7
//   elems 8..15 -> K = 8*(lane>>4) + 16..23
__device__ __forceinline__ v16bf load_a_frag(const __bf16* __restrict__ A, int lda, int lane) {
  int m  = lane & 15;
  int kb = (lane >> 4) << 3;
  const __bf16* p = A + (size_t)m * lda + kb;
  union { v16bf v; v8bf h[2]; } u;
  u.h[0] = *(const v8bf*)(p);
  u.h[1] = *(const v8bf*)(p + 16);
  return u.v;
}
// B fragment (32x16 bf16): lane n = lane&15, same K order as A fragment.
// B[k,n] = W[n,k] (weights row-major over K) -> 8 contiguous bf16 at kb and 8 at kb+16 from W row n.
__device__ __forceinline__ v16bf load_b_frag(const __bf16* __restrict__ W, int ldw, int lane) {
  int n  = lane & 15;
  int kb = (lane >> 4) << 3;
  const __bf16* p = W + (size_t)n * ldw + kb;
  union { v16bf v; v8bf h[2]; } u;
  u.h[0] = *(const v8bf*)(p);
  u.h[1] = *(const v8bf*)(p + 16);
  return u.v;
}

// 32(M) x 64(N) tile per wave: 2 A frags x 4 B frags -> 8 WMMAs / k-step.
__global__ void __launch_bounds__(32) k_gemm_bf16(
    const __bf16* __restrict__ A, const __bf16* __restrict__ W,
    const float* __restrict__ bias, float* __restrict__ C,
    int M, int Nc, int K) {
  int m0 = blockIdx.x * 32;
  int n0 = blockIdx.y * 64;
  int lane = threadIdx.x;

  v8f c00 = {}, c01 = {}, c02 = {}, c03 = {};
  v8f c10 = {}, c11 = {}, c12 = {}, c13 = {};
  const __bf16* A0 = A + (size_t)m0 * K;
  const __bf16* A1 = A + (size_t)(m0 + 16) * K;
  const __bf16* W0 = W + (size_t)(n0     ) * K;
  const __bf16* W1 = W + (size_t)(n0 + 16) * K;
  const __bf16* W2 = W + (size_t)(n0 + 32) * K;
  const __bf16* W3 = W + (size_t)(n0 + 48) * K;

  for (int k0 = 0; k0 < K; k0 += 32) {
    v16bf a0 = load_a_frag(A0 + k0, K, lane);
    v16bf a1 = load_a_frag(A1 + k0, K, lane);
    v16bf b0 = load_b_frag(W0 + k0, K, lane);
    v16bf b1 = load_b_frag(W1 + k0, K, lane);
    v16bf b2 = load_b_frag(W2 + k0, K, lane);
    v16bf b3 = load_b_frag(W3 + k0, K, lane);
    c00 = __builtin_amdgcn_wmma_f32_16x16x32_bf16(false, a0, false, b0, (short)0, c00, false, false);
    c10 = __builtin_amdgcn_wmma_f32_16x16x32_bf16(false, a1, false, b0, (short)0, c10, false, false);
    c01 = __builtin_amdgcn_wmma_f32_16x16x32_bf16(false, a0, false, b1, (short)0, c01, false, false);
    c11 = __builtin_amdgcn_wmma_f32_16x16x32_bf16(false, a1, false, b1, (short)0, c11, false, false);
    c02 = __builtin_amdgcn_wmma_f32_16x16x32_bf16(false, a0, false, b2, (short)0, c02, false, false);
    c12 = __builtin_amdgcn_wmma_f32_16x16x32_bf16(false, a1, false, b2, (short)0, c12, false, false);
    c03 = __builtin_amdgcn_wmma_f32_16x16x32_bf16(false, a0, false, b3, (short)0, c03, false, false);
    c13 = __builtin_amdgcn_wmma_f32_16x16x32_bf16(false, a1, false, b3, (short)0, c13, false, false);
  }

  int nL = lane & 15;
  int mB = (lane >> 4) * 8;
  float bb0 = bias[n0 + nL];
  float bb1 = bias[n0 + 16 + nL];
  float bb2 = bias[n0 + 32 + nL];
  float bb3 = bias[n0 + 48 + nL];
#pragma unroll
  for (int r = 0; r < 8; ++r) {
    int ma = m0 + mB + r;
    if (ma < M) {
      float* crow = C + (size_t)ma * Nc;
      crow[n0 + nL]      = c00[r] + bb0;
      crow[n0 + 16 + nL] = c01[r] + bb1;
      crow[n0 + 32 + nL] = c02[r] + bb2;
      crow[n0 + 48 + nL] = c03[r] + bb3;
    }
    int mb = m0 + 16 + mB + r;
    if (mb < M) {
      float* crow = C + (size_t)mb * Nc;
      crow[n0 + nL]      = c10[r] + bb0;
      crow[n0 + 16 + nL] = c11[r] + bb1;
      crow[n0 + 32 + nL] = c12[r] + bb2;
      crow[n0 + 48 + nL] = c13[r] + bb3;
    }
  }
}

// ------------------------------------------------------------------- RMSNorm
__global__ void __launch_bounds__(256) k_rmsnorm(
    const float* __restrict__ Y, const float* __restrict__ lnw,
    __bf16* __restrict__ Yn, int Dd) {
  int m = blockIdx.x;
  const float* y = Y + (size_t)m * Dd;
  __shared__ float red[256];
  float acc = 0.f;
  for (int d = threadIdx.x; d < Dd; d += 256) { float v = y[d]; acc += v * v; }
  red[threadIdx.x] = acc;
  __syncthreads();
  for (int s = 128; s > 0; s >>= 1) {
    if (threadIdx.x < s) red[threadIdx.x] += red[threadIdx.x + s];
    __syncthreads();
  }
  float r = rsqrtf(red[0] / (float)Dd + 1e-6f);
  for (int d = threadIdx.x; d < Dd; d += 256)
    Yn[(size_t)m * Dd + d] = (__bf16)(lnw[d] * y[d] * r);
}

// ------------------------------------------------------------------- tail outputs
__global__ void k_tail(const int* __restrict__ ntt, const float* __restrict__ sums,
                       float* __restrict__ out, int B) {
  int b = threadIdx.x;
  if (b < B) { out[b] = (float)ntt[b]; out[B + b] = sums[b]; }
}

extern "C" void kernel_launch(void* const* d_in, const int* in_sizes, int n_in,
                              void* d_out, int out_size, void* d_ws, size_t ws_size,
                              hipStream_t stream) {
  const float* audio  = (const float*)d_in[0];
  const float* cif_w  = (const float*)d_in[1];
  const float* cif_b  = (const float*)d_in[2];
  const float* ln_w   = (const float*)d_in[3];
  const float* text_w = (const float*)d_in[4];
  const float* text_b = (const float*)d_in[5];
  const int* num_frames = (const int*)d_in[6];
  const int* num_text   = (const int*)d_in[7];
  // d_in[8] = max_num_tokens (device scalar); N derived from out_size instead.

  int B    = in_sizes[6];                       // 16
  int Dd   = in_sizes[3];                       // 1024 (ln_w length)
  int T    = in_sizes[0] / (B * Dd);            // 1500
  int Dtxt = in_sizes[4] / Dd;                  // 4096
  int N    = (out_size - 2 * B) / (B * Dtxt);   // 375
  int M    = B * N;                             // 6000
  int Mp   = (M + 31) & ~31;                    // padded rows for 32-row tiles
  int K1   = Dd - 1;                            // 1023 (hidden feature dim)
  int Kp   = Dd;                                // padded K = 1024

  // workspace layout (~61 MB total)
  size_t off = 0;
  auto alloc = [&](size_t bytes) -> void* {
    void* p = (char*)d_ws + off;
    off = (off + bytes + 255) & ~(size_t)255;
    return p;
  };
  float*  alphas    = (float*)alloc((size_t)B * T * 4);
  float*  sums      = (float*)alloc((size_t)B * 4);
  float*  ev_w      = (float*)alloc((size_t)B * EV_CAP * 4);
  int*    ev_t      = (int*)  alloc((size_t)B * EV_CAP * 4);
  int*    tok_start = (int*)  alloc((size_t)B * (N + 1) * 4);
  __bf16* Hbf       = (__bf16*)alloc((size_t)Mp * Kp * 2);
  __bf16* W1bf      = (__bf16*)alloc((size_t)Dd * Kp * 2);
  __bf16* W2bf      = (__bf16*)alloc((size_t)Dtxt * Kp * 2);
  float*  Y1        = (float*)alloc((size_t)Mp * Dd * 4);
  __bf16* Ynbf      = (__bf16*)alloc((size_t)Mp * Dd * 2);

  float* out = (float*)d_out;

  // 1) alphas + sums
  k_alphas<<<B, 256, 0, stream>>>(audio, num_frames, alphas, sums, T, Dd);
  // 2) sequential CIF -> sparse events (single wave; cross-batch any via ballot)
  k_cif_events<<<1, 32, 0, stream>>>(alphas, sums, num_text, ev_w, ev_t, tok_start, B, T, N);
  // 3) gather events -> H (bf16, K padded); zero the M..Mp pad rows (ws is poisoned)
  k_cif_gather<<<dim3(N, B), 256, 0, stream>>>(audio, ev_w, ev_t, tok_start, Hbf, T, Dd, N, Kp);
  if (Mp > M) {
    size_t padn = (size_t)(Mp - M) * Kp;
    unsigned blks = (unsigned)((padn + 255) / 256);
    k_zero_bf16<<<blks, 256, 0, stream>>>(Hbf + (size_t)M * Kp, padn);
    k_zero_bf16<<<blks, 256, 0, stream>>>(Ynbf + (size_t)M * Dd, padn);
  }
  // 4) weight conversion
  {
    size_t tot1 = (size_t)Dd * Kp;
    k_cvt_w<<<(unsigned)((tot1 + 255) / 256), 256, 0, stream>>>(cif_w, W1bf, Dd, K1, Kp);
    size_t tot2 = (size_t)Dtxt * Kp;
    k_cvt_w<<<(unsigned)((tot2 + 255) / 256), 256, 0, stream>>>(text_w, W2bf, Dtxt, Dd, Kp);
  }
  // 5) cif_proj: Y1[M,Dd] = H @ cif_w^T + cif_b   (WMMA bf16, 32x64 tile/wave)
  k_gemm_bf16<<<dim3(Mp / 32, Dd / 64), 32, 0, stream>>>(Hbf, W1bf, cif_b, Y1, M, Dd, Kp);
  // 6) RMSNorm -> bf16
  k_rmsnorm<<<M, 256, 0, stream>>>(Y1, ln_w, Ynbf, Dd);
  // 7) text_proj: out[M,Dtxt] = Yn @ text_w^T + text_b   (WMMA bf16)
  k_gemm_bf16<<<dim3(Mp / 32, Dtxt / 64), 32, 0, stream>>>(Ynbf, W2bf, text_b, out, M, Dtxt, Kp);
  // 8) scalar outputs
  k_tail<<<1, 32, 0, stream>>>(num_text, sums, out + (size_t)M * Dtxt, B);
}
